// AttentionDecoderRNN_89077621719306
// MI455X (gfx1250) — matrix operand order repaired and checked by reading, not compiled, not run on hardware
//
#include <hip/hip_runtime.h>
#include <hip/hip_bf16.h>
#include <math.h>

// ---------------- problem constants ----------------
#define Hc   256
#define Ac   64
#define Vc   32000
#define Tc   64
#define Bc   32
#define Sc   1024
#define G3H  (3 * Hc)          // 768
#define X2H  (2 * Hc)          // 512

// ---------------- WMMA types ----------------
typedef __attribute__((ext_vector_type(16))) __bf16 v16bf;
typedef __attribute__((ext_vector_type(8)))  float  v8f;

union Frag { v16bf v; uint4 q[2]; };

__device__ __forceinline__ unsigned short bfb(float f) {
  unsigned u = __builtin_bit_cast(unsigned, f);
  return (unsigned short)((u + 0x7FFFu + ((u >> 16) & 1u)) >> 16);  // RNE
}
__device__ __forceinline__ __bf16 f2bf(float f) {
  unsigned short h = bfb(f);
  return __builtin_bit_cast(__bf16, h);
}
__device__ __forceinline__ unsigned pk2(float a, float b) {
  return (unsigned)bfb(a) | ((unsigned)bfb(b) << 16);
}
__device__ __forceinline__ float sigmf(float x) { return 1.0f / (1.0f + expf(-x)); }

// ---------------- async global->LDS copy (gfx1250), guarded ----------------
#if defined(__has_builtin)
#  if __has_builtin(__builtin_amdgcn_global_load_async_to_lds_b128) && \
      __has_builtin(__builtin_amdgcn_s_wait_asynccnt)
#    define ASYNC_LDS 1
#  endif
#endif

#ifdef ASYNC_LDS
// builtin signature: void(v4i AS1* gaddr, v4i AS3* lds, Imm int off, Imm int cpol)
typedef int v4i_gcc __attribute__((vector_size(16)));
typedef __attribute__((address_space(1))) v4i_gcc as1_v4i;
typedef __attribute__((address_space(3))) v4i_gcc as3_v4i;
__device__ __forceinline__ void async_copy_b128(const void* g, void* l) {
  // AS1 ptr: generic addr == global addr.  AS3 ptr: LDS offset = addr[31:0].
  __builtin_amdgcn_global_load_async_to_lds_b128(
      (as1_v4i*)(size_t)g, (as3_v4i*)(unsigned)(size_t)l, 0, 0);
}
#endif

// ---------------------------------------------------------------------------
// f32 -> bf16 conversion, 4 elements/thread (b128 in, b64 out)
// ---------------------------------------------------------------------------
__global__ __launch_bounds__(256)
void f32_to_bf16_kernel(const float* __restrict__ src, __bf16* __restrict__ dst, int n4) {
  int i = blockIdx.x * blockDim.x + threadIdx.x;
  if (i >= n4) return;
  float4 f = reinterpret_cast<const float4*>(src)[i];
  uint2 o;
  o.x = pk2(f.x, f.y);
  o.y = pk2(f.z, f.w);
  reinterpret_cast<uint2*>(dst)[i] = o;
}

// ---------------------------------------------------------------------------
// Tiled bf16 WMMA GEMM:  Y[M,N] = act(X[M,K] @ W[N,K]^T + bias)
// block = 128 threads (4 waves); tile = 32x64; wave w owns a 32x16 strip.
// Double-buffered LDS: slab k+1 is copied (async-to-LDS when available)
// while WMMAs consume slab k; s_wait_asynccnt(3) retires the current slab.
// ACT: 0=none 1=relu 2=tanh ; OBF: bf16 output (else f32)
// ---------------------------------------------------------------------------
#define BM 32
#define BN 64
#define BK 32

template <int ACT, bool OBF>
__global__ __launch_bounds__(128)
void gemm_bf16_wmma(const __bf16* __restrict__ X, const __bf16* __restrict__ W,
                    const float* __restrict__ bias, void* __restrict__ Yv,
                    int M, int N, int K) {
  __shared__ uint4 XsQ[2][BM][5];   // 2 buffers x 32 rows x 4 chunks (+pad)
  __shared__ uint4 WsQ[2][BN][5];   // 2 buffers x 64 rows x 4 chunks (+pad)

  const int rowBase = blockIdx.x * BM;
  const int colBase = blockIdx.y * BN;
  const int tid  = threadIdx.x;
  const int wave = tid >> 5;
  const int lane = tid & 31;
  const int hf   = lane >> 4;
  const int l16  = lane & 15;

  // staging assignments (fixed per thread): 3 x b128 per slab
  const int xm  = tid >> 2,         xc  = tid & 3;
  const int wn0 = tid >> 2,         wc0 = tid & 3;
  const int wn1 = (tid + 128) >> 2, wc1 = tid & 3;

  const __bf16* xrow  = X + (size_t)(rowBase + xm) * K + xc * 8;
  const __bf16* wrow0 = W + (size_t)(colBase + wn0) * K + wc0 * 8;
  const __bf16* wrow1 = W + (size_t)(colBase + wn1) * K + wc1 * 8;

  auto stage = [&](int k0, int buf) {
#ifdef ASYNC_LDS
    async_copy_b128(xrow + k0,  &XsQ[buf][xm][xc]);
    async_copy_b128(wrow0 + k0, &WsQ[buf][wn0][wc0]);
    async_copy_b128(wrow1 + k0, &WsQ[buf][wn1][wc1]);
#else
    XsQ[buf][xm][xc]   = *reinterpret_cast<const uint4*>(xrow + k0);
    WsQ[buf][wn0][wc0] = *reinterpret_cast<const uint4*>(wrow0 + k0);
    WsQ[buf][wn1][wc1] = *reinterpret_cast<const uint4*>(wrow1 + k0);
    if (k0 + BK < K) {
      __builtin_prefetch(xrow + k0 + BK, 0, 1);
      __builtin_prefetch(wrow0 + k0 + BK, 0, 1);
      __builtin_prefetch(wrow1 + k0 + BK, 0, 1);
    }
#endif
  };

  v8f acc0 = {}, acc1 = {};

  stage(0, 0);
  int buf = 0;
  for (int k0 = 0; k0 < K; k0 += BK) {
    const bool hasNext = (k0 + BK < K);
    if (hasNext) stage(k0 + BK, buf ^ 1);
#ifdef ASYNC_LDS
    if (hasNext) __builtin_amdgcn_s_wait_asynccnt(3);  // slab k's 3 copies done
    else         __builtin_amdgcn_s_wait_asynccnt(0);
#endif
    __syncthreads();

    // A frag (16x32 bf16): elems 0-7 -> k=hf*8+i (chunk hf), 8-15 -> chunk 2+hf
    Frag a0, a1, bb;
    a0.q[0] = XsQ[buf][l16][hf];
    a0.q[1] = XsQ[buf][l16][2 + hf];
    a1.q[0] = XsQ[buf][16 + l16][hf];
    a1.q[1] = XsQ[buf][16 + l16][2 + hf];
    // B frag (32x16): elems 0-7 -> k=hf*8+i (chunk hf), 8-15 -> chunk 2+hf
    bb.q[0] = WsQ[buf][wave * 16 + l16][hf];
    bb.q[1] = WsQ[buf][wave * 16 + l16][2 + hf];

    acc0 = __builtin_amdgcn_wmma_f32_16x16x32_bf16(false, a0.v, false, bb.v,
                                                   (short)0, acc0, false, false);
    acc1 = __builtin_amdgcn_wmma_f32_16x16x32_bf16(false, a1.v, false, bb.v,
                                                   (short)0, acc1, false, false);
    __syncthreads();   // no wave may still read `buf` when it is re-staged
    buf ^= 1;
  }

  // C/D layout: m = j + hf*8, n = l16
  const int col = colBase + wave * 16 + l16;
  const float bv = bias ? bias[col] : 0.0f;
#pragma unroll
  for (int j = 0; j < 8; ++j) {
    int m0 = rowBase + hf * 8 + j;
    float v0 = acc0[j] + bv;
    float v1 = acc1[j] + bv;
    if (ACT == 1) { v0 = fmaxf(v0, 0.0f); v1 = fmaxf(v1, 0.0f); }
    if (ACT == 2) { v0 = tanhf(v0);       v1 = tanhf(v1); }
    if (OBF) {
      ((__bf16*)Yv)[(size_t)m0 * N + col]        = f2bf(v0);
      ((__bf16*)Yv)[(size_t)(m0 + 16) * N + col] = f2bf(v1);
    } else {
      ((float*)Yv)[(size_t)m0 * N + col]        = v0;
      ((float*)Yv)[(size_t)(m0 + 16) * N + col] = v1;
    }
  }
}

// ---------------------------------------------------------------------------
// fused recurrent-step kernels (block per batch element, latency-bound)
// ---------------------------------------------------------------------------
__global__ void zero_kernel(float* __restrict__ p, int n) {
  int i = blockIdx.x * blockDim.x + threadIdx.x;
  if (i < n) p[i] = 0.0f;
}

// full GRU cell for one timestep: gates + combine; writes h_new and Xcat row
__global__ __launch_bounds__(256)
void gru_step_kernel(const int* __restrict__ ytok, const float* __restrict__ emb,
                     float* __restrict__ hcur, const float* __restrict__ ctxc,
                     const float* __restrict__ w_ih, const float* __restrict__ b_ih,
                     const float* __restrict__ w_hh, const float* __restrict__ b_hh,
                     __bf16* __restrict__ xcat_t) {
  __shared__ float xs[X2H];
  __shared__ float hs[Hc];
  const int b = blockIdx.x, j = threadIdx.x;
  const int token = ytok[b];
  xs[j]      = emb[(size_t)token * Hc + j];
  xs[Hc + j] = ctxc[b * Hc + j];
  hs[j]      = hcur[b * Hc + j];
  __syncthreads();

  const float* wr = w_ih + (size_t)j * X2H;
  const float* wz = w_ih + (size_t)(Hc + j) * X2H;
  const float* wn = w_ih + (size_t)(2 * Hc + j) * X2H;
  float gr = b_ih[j], gz = b_ih[Hc + j], gn = b_ih[2 * Hc + j];
  for (int k = 0; k < X2H; ++k) {
    float x = xs[k];
    gr += x * wr[k]; gz += x * wz[k]; gn += x * wn[k];
  }
  const float* vr = w_hh + (size_t)j * Hc;
  const float* vz = w_hh + (size_t)(Hc + j) * Hc;
  const float* vn = w_hh + (size_t)(2 * Hc + j) * Hc;
  float hr = b_hh[j], hz = b_hh[Hc + j], hn = b_hh[2 * Hc + j];
  for (int k = 0; k < Hc; ++k) {
    float h = hs[k];
    hr += h * vr[k]; hz += h * vz[k]; hn += h * vn[k];
  }
  float r = sigmf(gr + hr);
  float z = sigmf(gz + hz);
  float n = tanhf(gn + r * hn);
  float hnew = (1.0f - z) * n + z * hs[j];
  hcur[b * Hc + j] = hnew;
  xcat_t[b * X2H + j] = f2bf(hnew);
}

// c1 = tanh(relu(h @ f1_w1^T + b1) @ f1_w2^T + b2), fused via LDS
__global__ __launch_bounds__(256)
void attn_mlp_kernel(const float* __restrict__ hcur,
                     const float* __restrict__ w1, const float* __restrict__ b1,
                     const float* __restrict__ w2, const float* __restrict__ b2,
                     float* __restrict__ c1) {
  __shared__ float hs[Hc], t1s[Hc];
  const int b = blockIdx.x, j = threadIdx.x;
  hs[j] = hcur[b * Hc + j];
  __syncthreads();
  const float* wr = w1 + (size_t)j * Hc;
  float acc = b1[j];
  for (int k = 0; k < Hc; ++k) acc += hs[k] * wr[k];
  t1s[j] = fmaxf(acc, 0.0f);
  __syncthreads();
  if (j < Ac) {
    const float* w2r = w2 + (size_t)j * Hc;
    float a = b2[j];
    for (int k = 0; k < Hc; ++k) a += t1s[k] * w2r[k];
    c1[b * Ac + j] = tanhf(a);
  }
}

// scores over S=1024 + softmax, fused in one block per batch element
__global__ __launch_bounds__(256)
void score_softmax_kernel(const float* __restrict__ c1, const float* __restrict__ c2,
                          float* __restrict__ att) {
  __shared__ float cs[Ac];
  __shared__ float red[256];
  const int b = blockIdx.x, tid = threadIdx.x;
  if (tid < Ac) cs[tid] = c1[b * Ac + tid];
  __syncthreads();
  float loc[4];
  float mx = -1e30f;
#pragma unroll
  for (int r = 0; r < 4; ++r) {
    int s = tid + r * 256;
    const float* row = c2 + ((size_t)s * Bc + b) * Ac;
    float acc = 0.0f;
    for (int a = 0; a < Ac; ++a) acc += cs[a] * row[a];
    loc[r] = acc;
    mx = fmaxf(mx, acc);
  }
  red[tid] = mx; __syncthreads();
  for (int st = 128; st > 0; st >>= 1) {
    if (tid < st) red[tid] = fmaxf(red[tid], red[tid + st]);
    __syncthreads();
  }
  float m = red[0]; __syncthreads();
  float sum = 0.0f;
#pragma unroll
  for (int r = 0; r < 4; ++r) { loc[r] = expf(loc[r] - m); sum += loc[r]; }
  red[tid] = sum; __syncthreads();
  for (int st = 128; st > 0; st >>= 1) {
    if (tid < st) red[tid] += red[tid + st];
    __syncthreads();
  }
  float inv = 1.0f / red[0];
#pragma unroll
  for (int r = 0; r < 4; ++r) att[b * Sc + tid + r * 256] = loc[r] * inv;
}

// ctx[b,:] = sum_s attw[b,s] * y_enc[s,b,:]; attw staged in LDS
__global__ __launch_bounds__(256)
void ctx_kernel(const float* __restrict__ att, const float* __restrict__ y_enc,
                float* __restrict__ ctxc, __bf16* __restrict__ xcat_t) {
  __shared__ float aw[Sc];
  const int b = blockIdx.x, hh = threadIdx.x;
#pragma unroll
  for (int r = 0; r < 4; ++r) aw[hh + r * 256] = att[b * Sc + hh + r * 256];
  __syncthreads();
  const float* ye = y_enc + (size_t)b * Hc + hh;
  float acc = 0.0f;
  for (int s = 0; s < Sc; ++s) acc += aw[s] * ye[(size_t)s * Bc * Hc];
  ctxc[b * Hc + hh] = acc;
  xcat_t[b * X2H + Hc + hh] = f2bf(acc);
}

// ---------------------------------------------------------------------------
// loss: single-pass online logsumexp per row (float4 loads), then final reduce
// ---------------------------------------------------------------------------
__global__ __launch_bounds__(256)
void nll_kernel(const float* __restrict__ preds, const int* __restrict__ y_out,
                float* __restrict__ nll, float* __restrict__ msk) {
  __shared__ float rm[256], rs[256];
  const int r = blockIdx.x, tid = threadIdx.x;
  const float4* row4 = reinterpret_cast<const float4*>(preds + (size_t)r * Vc);
  float m = -1e30f, s = 0.0f;
  for (int q = tid; q < Vc / 4; q += 256) {
    float4 f = row4[q];
    float lm = fmaxf(fmaxf(f.x, f.y), fmaxf(f.z, f.w));
    float nm = fmaxf(m, lm);
    s = s * expf(m - nm) + expf(f.x - nm) + expf(f.y - nm) + expf(f.z - nm) + expf(f.w - nm);
    m = nm;
  }
  rm[tid] = m; rs[tid] = s; __syncthreads();
  for (int st = 128; st > 0; st >>= 1) {
    if (tid < st) {
      float m2 = rm[tid + st], s2 = rs[tid + st];
      float nm = fmaxf(rm[tid], m2);
      rs[tid] = rs[tid] * expf(rm[tid] - nm) + s2 * expf(m2 - nm);
      rm[tid] = nm;
    }
    __syncthreads();
  }
  if (tid == 0) {
    int t = y_out[r];
    float lse = logf(rs[0]) + rm[0];
    float mk = (t != 0) ? 1.0f : 0.0f;
    const float* row = preds + (size_t)r * Vc;
    nll[r] = (lse - row[t]) * mk;
    msk[r] = mk;
  }
}

__global__ __launch_bounds__(256)
void loss_final_kernel(const float* __restrict__ nll, const float* __restrict__ msk,
                       float* __restrict__ out_loss) {
  __shared__ float rs[256], rc[256];
  int tid = threadIdx.x;
  float s = 0.0f, c = 0.0f;
  for (int r = tid; r < Tc * Bc; r += 256) { s += nll[r]; c += msk[r]; }
  rs[tid] = s; rc[tid] = c; __syncthreads();
  for (int st = 128; st > 0; st >>= 1) {
    if (tid < st) { rs[tid] += rs[tid + st]; rc[tid] += rc[tid + st]; }
    __syncthreads();
  }
  if (tid == 0) out_loss[0] = rs[0] / fmaxf(rc[0], 1.0f);
}

// ---------------------------------------------------------------------------
// host-side launch
// ---------------------------------------------------------------------------
extern "C" void kernel_launch(void* const* d_in, const int* in_sizes, int n_in,
                              void* d_out, int out_size, void* d_ws, size_t ws_size,
                              hipStream_t stream) {
  const int*   y_inputs  = (const int*)  d_in[0];
  const int*   y_outputs = (const int*)  d_in[1];
  const float* y_enc     = (const float*)d_in[2];
  const float* emb       = (const float*)d_in[4];
  const float* f1_w1     = (const float*)d_in[5];
  const float* f1_b1     = (const float*)d_in[6];
  const float* f1_w2     = (const float*)d_in[7];
  const float* f1_b2     = (const float*)d_in[8];
  const float* f2_w1     = (const float*)d_in[9];
  const float* f2_b1     = (const float*)d_in[10];
  const float* f2_w2     = (const float*)d_in[11];
  const float* f2_b2     = (const float*)d_in[12];
  const float* gru_w_ih  = (const float*)d_in[13];
  const float* gru_b_ih  = (const float*)d_in[14];
  const float* gru_w_hh  = (const float*)d_in[15];
  const float* gru_b_hh  = (const float*)d_in[16];
  const float* fc1_w     = (const float*)d_in[17];
  const float* fc1_b     = (const float*)d_in[18];
  const float* fc2_w     = (const float*)d_in[19];
  const float* fc2_b     = (const float*)d_in[20];

  float* preds = (float*)d_out;                        // [T*B, V]
  float* loss  = preds + (size_t)Tc * Bc * Vc;

  const size_t SB  = (size_t)Sc * Bc;                  // 32768
  const size_t MiB = 1024 * 1024;
  char* base = (char*)d_ws;

  // region A [0,16MiB): y_enc_bf during phase 1; Xcat/Y1/fc1_w bf16 afterwards
  __bf16* y_enc_bf = (__bf16*)(base);                  // 8,388,608 el
  __bf16* Xcat_bf  = (__bf16*)(base);                  // [T*B, 2H] (alias, phase 2+)
  __bf16* Y1_bf    = (__bf16*)(base + 2 * MiB);        // [T*B, H]
  __bf16* fc1_w_bf = (__bf16*)(base + 4 * MiB);        // 131,072 el
  // region B [16,32MiB): tmp1_bf during phase 1; fc2_w_bf afterwards
  __bf16* tmp1_bf  = (__bf16*)(base + 16 * MiB);       // [SB, H]
  __bf16* fc2_w_bf = (__bf16*)(base + 16 * MiB);       // 8,192,000 el (alias)
  // region C [32,33MiB): small bf16 weights
  __bf16* f2_w1_bf = (__bf16*)(base + 32 * MiB);       // 65,536 el
  __bf16* f2_w2_bf = (__bf16*)(base + 32 * MiB + 256 * 1024);  // 16,384 el
  // region D [33MiB..): f32 buffers
  float* c2   = (float*)(base + 33 * MiB);             // [SB, A] = 8 MiB
  float* fD   = (float*)(base + 41 * MiB + 64 * 1024);
  float* hcur = fD;                                    // [B, H]
  float* ctxc = hcur + Bc * Hc;                        // [B, H]
  float* c1   = ctxc + Bc * Hc;                        // [B, A]
  float* att  = c1 + Bc * Ac;                          // [B, S]
  float* nll  = att + (size_t)Bc * Sc;                 // [T*B]
  float* msk  = nll + Tc * Bc;                         // [T*B]

  // ---- phase 0: convert phase-1 operands to bf16, init state
  f32_to_bf16_kernel<<<(int)((SB * Hc / 4 + 255) / 256), 256, 0, stream>>>(
      y_enc, y_enc_bf, (int)(SB * Hc / 4));
  f32_to_bf16_kernel<<<(Hc * Hc / 4 + 255) / 256, 256, 0, stream>>>(
      f2_w1, f2_w1_bf, Hc * Hc / 4);
  f32_to_bf16_kernel<<<(Ac * Hc / 4 + 255) / 256, 256, 0, stream>>>(
      f2_w2, f2_w2_bf, Ac * Hc / 4);
  zero_kernel<<<(2 * Bc * Hc + 255) / 256, 256, 0, stream>>>(hcur, 2 * Bc * Hc);

  // ---- phase 1: loop-invariant encoder keys c2 (WMMA)
  gemm_bf16_wmma<1, true><<<dim3(SB / BM, Hc / BN), 128, 0, stream>>>(
      y_enc_bf, f2_w1_bf, f2_b1, tmp1_bf, (int)SB, Hc, Hc);
  gemm_bf16_wmma<2, false><<<dim3(SB / BM, Ac / BN), 128, 0, stream>>>(
      tmp1_bf, f2_w2_bf, f2_b2, c2, (int)SB, Ac, Hc);

  // phase-3 weights converted now that regions A/B are free
  f32_to_bf16_kernel<<<(Hc * X2H / 4 + 255) / 256, 256, 0, stream>>>(
      fc1_w, fc1_w_bf, Hc * X2H / 4);
  f32_to_bf16_kernel<<<(Vc * Hc / 4 + 255) / 256, 256, 0, stream>>>(
      fc2_w, fc2_w_bf, Vc * Hc / 4);

  // ---- phase 2: sequential recurrence, 4 fused launches per step
  for (int t = 0; t < Tc; ++t) {
    const int* ytok = y_inputs + t * Bc;
    __bf16* xcat_t = Xcat_bf + (size_t)t * Bc * X2H;
    gru_step_kernel<<<Bc, 256, 0, stream>>>(ytok, emb, hcur, ctxc,
                                            gru_w_ih, gru_b_ih, gru_w_hh, gru_b_hh, xcat_t);
    attn_mlp_kernel<<<Bc, 256, 0, stream>>>(hcur, f1_w1, f1_b1, f1_w2, f1_b2, c1);
    score_softmax_kernel<<<Bc, 256, 0, stream>>>(c1, c2, att);
    ctx_kernel<<<Bc, 256, 0, stream>>>(att, y_enc, ctxc, xcat_t);
  }

  // ---- phase 3: batched output head for all timesteps (WMMA)
  gemm_bf16_wmma<1, true><<<dim3((Tc * Bc) / BM, Hc / BN), 128, 0, stream>>>(
      Xcat_bf, fc1_w_bf, fc1_b, Y1_bf, Tc * Bc, Hc, X2H);
  gemm_bf16_wmma<0, false><<<dim3((Tc * Bc) / BM, Vc / BN), 128, 0, stream>>>(
      Y1_bf, fc2_w_bf, fc2_b, preds, Tc * Bc, Vc, Hc);

  // ---- phase 4: masked mean cross-entropy (deterministic)
  nll_kernel<<<Tc * Bc, 256, 0, stream>>>(preds, y_outputs, nll, msk);
  loss_final_kernel<<<1, 256, 0, stream>>>(nll, msk, loss);
}
